// LiquidModel_7258494730506
// MI455X (gfx1250) — matrix operand order repaired and checked by reading, not compiled, not run on hardware
//
#include <hip/hip_runtime.h>
#include <hip/hip_bf16.h>

typedef __bf16 bf16;
typedef __attribute__((ext_vector_type(16))) __bf16 v16bf;
typedef __attribute__((ext_vector_type(8)))  __bf16 v8bf;
typedef __attribute__((ext_vector_type(8)))  float  v8f;

union BF16x16 { v16bf v; v8bf h[2]; bf16 s[16]; };

// ---------------------------------------------------------------------------
// Gating: logits for token 0 only (softmax monotonic -> top-k of logits),
// pick top-2 expert indices like lax.top_k (first occurrence wins).
// ---------------------------------------------------------------------------
__global__ __launch_bounds__(32)
void gate_topk_kernel(const float* __restrict__ x, const float* __restrict__ gw,
                      const float* __restrict__ gb, int* __restrict__ sel)
{
    __shared__ float logits[8];
    const int e = threadIdx.x;
    if (e < 8) {
        float acc = gb[e];
        for (int i = 0; i < 1024; ++i) acc += x[i] * gw[i * 8 + e];
        logits[e] = acc;
    }
    __syncthreads();
    if (threadIdx.x == 0) {
        int i0 = 0; float v0 = logits[0];
        for (int k = 1; k < 8; ++k) if (logits[k] > v0) { v0 = logits[k]; i0 = k; }
        int i1 = (i0 == 0) ? 1 : 0; float v1 = logits[i1];
        for (int k = 0; k < 8; ++k)
            if (k != i0 && logits[k] > v1) { v1 = logits[k]; i1 = k; }
        sel[0] = i0; sel[1] = i1;
    }
}

// ---------------------------------------------------------------------------
// Average the two selected expert weights -> bf16, biases -> f32.
// ---------------------------------------------------------------------------
__global__ __launch_bounds__(256)
void avg_expert_kernel(const float* __restrict__ ew_l, const float* __restrict__ eb_l,
                       const int* __restrict__ sel, bf16* __restrict__ wout,
                       float* __restrict__ bout)
{
    const long gid = (long)blockIdx.x * 256 + threadIdx.x;
    const long s0 = sel[0], s1 = sel[1];
    const float* W0 = ew_l + s0 * (1024L * 1024);
    const float* W1 = ew_l + s1 * (1024L * 1024);
    const long i = gid * 4;
    const float4 a = *(const float4*)(W0 + i);
    const float4 b = *(const float4*)(W1 + i);
    wout[i + 0] = (bf16)(0.5f * (a.x + b.x));
    wout[i + 1] = (bf16)(0.5f * (a.y + b.y));
    wout[i + 2] = (bf16)(0.5f * (a.z + b.z));
    wout[i + 3] = (bf16)(0.5f * (a.w + b.w));
    if (gid < 1024)
        bout[gid] = 0.5f * (eb_l[s0 * 1024 + gid] + eb_l[s1 * 1024 + gid]);
}

// ---------------------------------------------------------------------------
// f32 -> bf16 (n multiple of 1024). Used for the input x and weights only.
// ---------------------------------------------------------------------------
__global__ __launch_bounds__(256)
void f32_to_bf16_kernel(const float* __restrict__ in, bf16* __restrict__ out, long n)
{
    const long i = ((long)blockIdx.x * 256 + threadIdx.x) * 4;
    if (i + 3 < n) {
        const float4 f = *(const float4*)(in + i);
        out[i + 0] = (bf16)f.x;
        out[i + 1] = (bf16)f.y;
        out[i + 2] = (bf16)f.z;
        out[i + 3] = (bf16)f.w;
    }
}

// ---------------------------------------------------------------------------
// Tiled WMMA bf16 GEMM, double-buffered LDS, 128x128 block tile, BK=32.
// A tile staged with GLOBAL_LOAD_ASYNC_TO_LDS_B128 (ASYNCcnt path, no data
// VGPRs); B tile staged via VGPR transpose. 8 wave32s, each computing 32x64
// (2x4 WMMA tiles). Optional f32/bf16 fused epilogue, bias + optional ReLU.
// ---------------------------------------------------------------------------
template<int ACT>
__global__ __launch_bounds__(256)
void gemm_bf16_kernel(const bf16* __restrict__ A, const bf16* __restrict__ B,
                      const float* __restrict__ bias, float* __restrict__ C,
                      bf16* __restrict__ Cb, int M, int N, int K)
{
    __shared__ bf16 As[2][128][40];   // 80B row stride (16B-aligned)
    __shared__ bf16 Bst[2][128][40];  // [n][k] transposed

    const int tid   = threadIdx.x;
    const int lane  = tid & 31;
    const int wave  = tid >> 5;
    const int lrow  = lane & 15;
    const int khalf = lane >> 4;
    const int waveM = wave >> 1;   // 0..3
    const int waveN = wave & 1;    // 0..1

    const int m0 = blockIdx.y * 128;
    const int n0 = blockIdx.x * 128;

    const int arow = tid >> 1;         // 0..127
    const int acol = (tid & 1) * 16;   // 0 or 16
    const int bkk  = tid >> 3;         // 0..31
    const int bnn  = (tid & 7) * 16;   // 0..112

    // A tile: async memory->LDS copy (two B128 per thread, offset:16 advances
    // both the global address and the LDS address).
    auto stageA = [&](int buf, int bk) {
        const unsigned long long ga =
            (unsigned long long)(A + (long)(m0 + arow) * K + bk + acol);
        const unsigned lds =
            (unsigned)(unsigned long long)&As[buf][arow][acol];
        asm volatile("global_load_async_to_lds_b128 %0, %1, off"
                     :: "v"(lds), "v"(ga) : "memory");
        asm volatile("global_load_async_to_lds_b128 %0, %1, off offset:16"
                     :: "v"(lds), "v"(ga) : "memory");
    };
    // B tile: load 16 contiguous bf16 per thread, store transposed.
    auto stageB = [&](int buf, int bk) {
        const bf16* bp = B + (long)(bk + bkk) * N + n0 + bnn;
        const v8bf b0 = *(const v8bf*)bp;
        const v8bf b1 = *(const v8bf*)(bp + 8);
#pragma unroll
        for (int e = 0; e < 8; ++e) Bst[buf][bnn + e][bkk] = b0[e];
#pragma unroll
        for (int e = 0; e < 8; ++e) Bst[buf][bnn + 8 + e][bkk] = b1[e];
    };

    const v8f vzero = {0.f, 0.f, 0.f, 0.f, 0.f, 0.f, 0.f, 0.f};
    v8f acc[2][4];
#pragma unroll
    for (int ti = 0; ti < 2; ++ti)
#pragma unroll
        for (int tj = 0; tj < 4; ++tj) acc[ti][tj] = vzero;

    stageA(0, 0);
    stageB(0, 0);
    asm volatile("s_wait_asynccnt 0" ::: "memory");
    __syncthreads();

    int buf = 0;
    for (int bk = 0; bk < K; bk += 32) {
        if (bk + 32 < K) {
            stageA(buf ^ 1, bk + 32);
            stageB(buf ^ 1, bk + 32);
            if (bk + 64 < K) {
                __builtin_prefetch(A + (long)(m0 + arow) * K + bk + 64 + acol, 0, 1);
                __builtin_prefetch(B + (long)(bk + 64 + bkk) * N + n0 + bnn, 0, 1);
            }
        }

        BF16x16 af[2], bfg[4];
#pragma unroll
        for (int ti = 0; ti < 2; ++ti) {
            const bf16* p = &As[buf][waveM * 32 + ti * 16 + lrow][khalf * 8];
            af[ti].h[0] = *(const v8bf*)p;        // K = khalf*8 + 0..7
            af[ti].h[1] = *(const v8bf*)(p + 16); // K = 16 + khalf*8 + 0..7
        }
#pragma unroll
        for (int tj = 0; tj < 4; ++tj) {
            const bf16* p = &Bst[buf][waveN * 64 + tj * 16 + lrow][khalf * 8];
            bfg[tj].h[0] = *(const v8bf*)p;       // K = khalf*8 + 0..7
            bfg[tj].h[1] = *(const v8bf*)(p + 16); // K = 16 + khalf*8 + 0..7
        }
#pragma unroll
        for (int ti = 0; ti < 2; ++ti)
#pragma unroll
            for (int tj = 0; tj < 4; ++tj)
                acc[ti][tj] = __builtin_amdgcn_wmma_f32_16x16x32_bf16(
                    false, af[ti].v, false, bfg[tj].v, (short)0, acc[ti][tj],
                    false, false);

        asm volatile("s_wait_asynccnt 0" ::: "memory");
        __syncthreads();
        buf ^= 1;
    }

#pragma unroll
    for (int ti = 0; ti < 2; ++ti) {
        const int crow = m0 + waveM * 32 + ti * 16 + khalf * 8;
#pragma unroll
        for (int tj = 0; tj < 4; ++tj) {
            const int ccol = n0 + waveN * 64 + tj * 16 + lrow;
            const float bv = bias[ccol];
#pragma unroll
            for (int r = 0; r < 8; ++r) {
                float v = acc[ti][tj][r] + bv;
                if (ACT) v = fmaxf(v, 0.f);
                if (C)  C[(long)(crow + r) * N + ccol] = v;
                if (Cb) Cb[(long)(crow + r) * N + ccol] = (bf16)v;
            }
        }
    }
}

// ---------------------------------------------------------------------------
// Transpose V out of qkv (bf16 [4096,3072]) into Vt[h][dh][N] (bf16).
// 32x32 tiles via LDS; grid (128, 8, 4), block 256.
// ---------------------------------------------------------------------------
__global__ __launch_bounds__(256)
void v_transpose_kernel(const bf16* __restrict__ qkv, bf16* __restrict__ vt)
{
    __shared__ bf16 tile[32][33];
    const int tx = threadIdx.x & 31;
    const int ty = threadIdx.x >> 5;
    const int h  = blockIdx.z;
    const int n0 = blockIdx.x * 32;
    const int d0 = blockIdx.y * 32;
#pragma unroll
    for (int i = 0; i < 4; ++i)
        tile[ty + i * 8][tx] =
            qkv[(long)(n0 + ty + i * 8) * 3072 + 2048 + h * 256 + d0 + tx];
    __syncthreads();
#pragma unroll
    for (int i = 0; i < 4; ++i)
        vt[(long)(h * 256 + d0 + ty + i * 8) * 4096 + n0 + tx] = tile[tx][ty + i * 8];
}

// ---------------------------------------------------------------------------
// Flash attention. qkv bf16 [4096,3072] (Q at h*256, K at 1024+h*256),
// vt bf16 [4][256][4096]. One wave per 16 queries per head; online softmax;
// output written directly as bf16 [4096,1024].
// ---------------------------------------------------------------------------
__global__ __launch_bounds__(32)
void flash_attn_kernel(const bf16* __restrict__ qkv, const bf16* __restrict__ vt,
                       bf16* __restrict__ outb)
{
    __shared__ bf16 Plds[16][32];

    const int lane  = threadIdx.x & 31;
    const int lrow  = lane & 15;
    const int khalf = lane >> 4;
    const int h  = blockIdx.y;
    const int q0 = blockIdx.x * 16;

    const bf16* Qb = qkv + h * 256;
    const bf16* Kb = qkv + 1024 + h * 256;
    const bf16* Vt = vt + (long)h * 256 * 4096;

    BF16x16 qf[8];
#pragma unroll
    for (int c = 0; c < 8; ++c) {
        const bf16* p = Qb + (long)(q0 + lrow) * 3072 + c * 32 + khalf * 8;
        qf[c].h[0] = *(const v8bf*)p;
        qf[c].h[1] = *(const v8bf*)(p + 16);
    }

    const v8f vzero = {0.f, 0.f, 0.f, 0.f, 0.f, 0.f, 0.f, 0.f};
    v8f o[16];
#pragma unroll
    for (int t = 0; t < 16; ++t) o[t] = vzero;
    float m[8], l[8];
#pragma unroll
    for (int r = 0; r < 8; ++r) { m[r] = -3.0e38f; l[r] = 0.f; }

    for (int j = 0; j < 4096; j += 32) {
        // S = Q @ K^T for 32 keys (two 16x16 tiles), K streamed from global
        v8f s0 = vzero, s1 = vzero;
#pragma unroll
        for (int c = 0; c < 8; ++c) {
            BF16x16 kf0, kf1;
            const bf16* p0 = Kb + (long)(j + lrow) * 3072 + c * 32 + khalf * 8;
            kf0.h[0] = *(const v8bf*)p0;
            kf0.h[1] = *(const v8bf*)(p0 + 16);
            const bf16* p1 = Kb + (long)(j + 16 + lrow) * 3072 + c * 32 + khalf * 8;
            kf1.h[0] = *(const v8bf*)p1;
            kf1.h[1] = *(const v8bf*)(p1 + 16);
            s0 = __builtin_amdgcn_wmma_f32_16x16x32_bf16(
                false, qf[c].v, false, kf0.v, (short)0, s0, false, false);
            s1 = __builtin_amdgcn_wmma_f32_16x16x32_bf16(
                false, qf[c].v, false, kf1.v, (short)0, s1, false, false);
        }
        s0 = s0 * 0.0625f;  // 1/sqrt(256)
        s1 = s1 * 0.0625f;

        // online softmax; C-layout row = khalf*8 + r, col = lrow (+16 for s1)
#pragma unroll
        for (int r = 0; r < 8; ++r) {
            float mx = fmaxf(s0[r], s1[r]);
            mx = fmaxf(mx, __shfl_xor(mx, 1));
            mx = fmaxf(mx, __shfl_xor(mx, 2));
            mx = fmaxf(mx, __shfl_xor(mx, 4));
            mx = fmaxf(mx, __shfl_xor(mx, 8));
            const float mnew  = fmaxf(m[r], mx);
            const float alpha = __expf(m[r] - mnew);
            const float p0 = __expf(s0[r] - mnew);
            const float p1 = __expf(s1[r] - mnew);
            float sum = p0 + p1;
            sum += __shfl_xor(sum, 1);
            sum += __shfl_xor(sum, 2);
            sum += __shfl_xor(sum, 4);
            sum += __shfl_xor(sum, 8);
            l[r] = l[r] * alpha + sum;
            m[r] = mnew;
#pragma unroll
            for (int t = 0; t < 16; ++t) o[t][r] = o[t][r] * alpha;
            Plds[khalf * 8 + r][lrow]      = (bf16)p0;
            Plds[khalf * 8 + r][16 + lrow] = (bf16)p1;
        }
        __syncthreads();

        // P fragment (A layout 16x32) via LDS transpose
        BF16x16 pf;
        {
            const bf16* pp = &Plds[lrow][khalf * 8];
            pf.h[0] = *(const v8bf*)pp;
            pf.h[1] = *(const v8bf*)(pp + 16);
        }
        // O += P @ V: V fragments are contiguous b128 loads from Vt
#pragma unroll
        for (int t = 0; t < 16; ++t) {
            BF16x16 vf;
            const bf16* vp = Vt + (long)(t * 16 + lrow) * 4096 + j + khalf * 8;
            vf.h[0] = *(const v8bf*)vp;
            vf.h[1] = *(const v8bf*)(vp + 16);
            o[t] = __builtin_amdgcn_wmma_f32_16x16x32_bf16(
                false, pf.v, false, vf.v, (short)0, o[t], false, false);
        }
        __syncthreads();
    }

#pragma unroll
    for (int r = 0; r < 8; ++r) {
        const float inv = 1.0f / l[r];
#pragma unroll
        for (int t = 0; t < 16; ++t)
            outb[(long)(q0 + khalf * 8 + r) * 1024 + h * 256 + t * 16 + lrow] =
                (bf16)(o[t][r] * inv);
    }
}

// ---------------------------------------------------------------------------
// out = LayerNorm(x + d) * g + b (rows of 1024); fused f32 + bf16 outputs.
// ---------------------------------------------------------------------------
__global__ __launch_bounds__(256)
void ln_residual_kernel(const float* __restrict__ x, const float* __restrict__ d,
                        const float* __restrict__ g, const float* __restrict__ b,
                        float* __restrict__ out, bf16* __restrict__ outb)
{
    __shared__ float red[256];
    const int row = blockIdx.x;
    const int tid = threadIdx.x;
    float v[4];
    float s = 0.f;
#pragma unroll
    for (int i = 0; i < 4; ++i) {
        const int c = tid * 4 + i;
        v[i] = x[(long)row * 1024 + c] + d[(long)row * 1024 + c];
        s += v[i];
    }
    red[tid] = s;
    __syncthreads();
    for (int st = 128; st > 0; st >>= 1) {
        if (tid < st) red[tid] += red[tid + st];
        __syncthreads();
    }
    const float mu = red[0] * (1.f / 1024.f);
    __syncthreads();
    float s2 = 0.f;
#pragma unroll
    for (int i = 0; i < 4; ++i) { const float t = v[i] - mu; s2 += t * t; }
    red[tid] = s2;
    __syncthreads();
    for (int st = 128; st > 0; st >>= 1) {
        if (tid < st) red[tid] += red[tid + st];
        __syncthreads();
    }
    const float rstd = rsqrtf(red[0] * (1.f / 1024.f) + 1e-5f);
#pragma unroll
    for (int i = 0; i < 4; ++i) {
        const int c = tid * 4 + i;
        const float r = (v[i] - mu) * rstd * g[c] + b[c];
        out[(long)row * 1024 + c]  = r;
        outb[(long)row * 1024 + c] = (bf16)r;
    }
}

// ---------------------------------------------------------------------------
// Host-side orchestration
// ---------------------------------------------------------------------------
static void conv_launch(const float* in, bf16* out, long n, hipStream_t s)
{
    f32_to_bf16_kernel<<<(int)(n / 1024), 256, 0, s>>>(in, out, n);
}

static void gemm_launch(const bf16* A, const bf16* B, const float* bias,
                        float* C, bf16* Cb, int M, int N, int K, int act,
                        hipStream_t s)
{
    dim3 grid(N / 128, M / 128);
    if (act) gemm_bf16_kernel<1><<<grid, 256, 0, s>>>(A, B, bias, C, Cb, M, N, K);
    else     gemm_bf16_kernel<0><<<grid, 256, 0, s>>>(A, B, bias, C, Cb, M, N, K);
}

extern "C" void kernel_launch(void* const* d_in, const int* in_sizes, int n_in,
                              void* d_out, int out_size, void* d_ws, size_t ws_size,
                              hipStream_t stream)
{
    (void)in_sizes; (void)n_in; (void)out_size; (void)ws_size;
    const float* x    = (const float*)d_in[0];
    const float* gw   = (const float*)d_in[1];
    const float* gb   = (const float*)d_in[2];
    const float* ew   = (const float*)d_in[3];
    const float* eb   = (const float*)d_in[4];
    const float* qkvw = (const float*)d_in[5];
    const float* qkvb = (const float*)d_in[6];
    const float* ow   = (const float*)d_in[7];
    const float* ob   = (const float*)d_in[8];
    const float* ln1g = (const float*)d_in[9];
    const float* ln1b = (const float*)d_in[10];
    const float* ln2g = (const float*)d_in[11];
    const float* ln2b = (const float*)d_in[12];
    const float* f1w  = (const float*)d_in[13];
    const float* f1b  = (const float*)d_in[14];
    const float* f2w  = (const float*)d_in[15];
    const float* f2b  = (const float*)d_in[16];
    const float* ffw  = (const float*)d_in[17];
    const float* ffb  = (const float*)d_in[18];
    const float* cfw  = (const float*)d_in[19];
    const float* cfb  = (const float*)d_in[20];
    const float* k1w  = (const float*)d_in[21];
    const float* k1b  = (const float*)d_in[22];
    const float* k2w  = (const float*)d_in[23];
    const float* k2b  = (const float*)d_in[24];
    const float* outw = (const float*)d_in[25];
    const float* outb = (const float*)d_in[26];

    char* ws = (char*)d_ws;
    size_t off = 0;
    auto take = [&](size_t bytes) -> char* {
        char* p = ws + off;
        off += (bytes + 255) & ~(size_t)255;
        return p;
    };
    int*   sel  = (int*)  take(256);
    float* bavg = (float*)take(1024 * 4);
    float* xA   = (float*)take(4096L * 1024 * 4);     // f32 ping
    float* xB   = (float*)take(4096L * 1024 * 4);     // f32 pong
    float* xC   = (float*)take(4096L * 1024 * 4);     // f32 third
    bf16*  hA   = (bf16*) take(4096L * 3072 * 2);     // bf16 big (qkv / ff mid)
    bf16*  hB   = (bf16*) take(4096L * 1024 * 2);     // bf16 ping
    bf16*  hC   = (bf16*) take(4096L * 1024 * 2);     // bf16 pong / Vt
    bf16*  wbf  = (bf16*) take(3072L * 1024 * 2);     // bf16 weight staging
    // total ~94 MB of workspace (fits the 192 MB L2)

    // ---- MoE layers (expert selection from token 0, applied to all) ----
    conv_launch(x, hA, 4096L * 1024, stream);
    gate_topk_kernel<<<1, 32, 0, stream>>>(x, gw, gb, sel);
    avg_expert_kernel<<<1024, 256, 0, stream>>>(ew, eb, sel, wbf, bavg);
    gemm_launch(hA, wbf, bavg, xA, hB, 4096, 1024, 1024, 0, stream);

    gate_topk_kernel<<<1, 32, 0, stream>>>(xA, gw + 1024L * 8, gb + 8, sel);
    avg_expert_kernel<<<1024, 256, 0, stream>>>(ew + 8L * 1024 * 1024,
                                                eb + 8L * 1024, sel, wbf, bavg);
    gemm_launch(hB, wbf, bavg, xB, hC, 4096, 1024, 1024, 0, stream);

    gate_topk_kernel<<<1, 32, 0, stream>>>(xB, gw + 2L * 1024 * 8, gb + 16, sel);
    avg_expert_kernel<<<1024, 256, 0, stream>>>(ew + 16L * 1024 * 1024,
                                                eb + 16L * 1024, sel, wbf, bavg);
    gemm_launch(hC, wbf, bavg, xA, hB, 4096, 1024, 1024, 0, stream);
    // xA = x0 (f32 residual source), hB = x0 (bf16)

    // ---- Transformer encoder layer ----
    conv_launch(qkvw, wbf, 1024L * 3072, stream);
    gemm_launch(hB, wbf, qkvb, nullptr, hA, 4096, 3072, 1024, 0, stream); // qkv bf16
    {
        dim3 grid(128, 8, 4);
        v_transpose_kernel<<<grid, 256, 0, stream>>>(hA, hC);             // Vt
    }
    {
        dim3 grid(256, 4);
        flash_attn_kernel<<<grid, 32, 0, stream>>>(hA, hC, hB);           // attn bf16
    }
    conv_launch(ow, wbf, 1024L * 1024, stream);
    gemm_launch(hB, wbf, ob, xB, nullptr, 4096, 1024, 1024, 0, stream);   // o-proj f32
    ln_residual_kernel<<<4096, 256, 0, stream>>>(xA, xB, ln1g, ln1b, xC, hC);

    conv_launch(f1w, wbf, 1024L * 2048, stream);
    gemm_launch(hC, wbf, f1b, nullptr, hA, 4096, 2048, 1024, 1, stream);  // relu
    conv_launch(f2w, wbf, 2048L * 1024, stream);
    gemm_launch(hA, wbf, f2b, xB, nullptr, 4096, 1024, 2048, 0, stream);
    ln_residual_kernel<<<4096, 256, 0, stream>>>(xC, xB, ln2g, ln2b, xA, hB);

    // ---- Trailing dense stack ----
    conv_launch(ffw, wbf, 1024L * 1024, stream);
    gemm_launch(hB, wbf, ffb, nullptr, hC, 4096, 1024, 1024, 0, stream);

    conv_launch(cfw, wbf, 1024L * 1024, stream);
    gemm_launch(hC, wbf, cfb, nullptr, hB, 4096, 1024, 1024, 0, stream);

    conv_launch(k1w, wbf, 1024L * 1024, stream);
    gemm_launch(hB, wbf, k1b, nullptr, hC, 4096, 1024, 1024, 1, stream);  // relu

    conv_launch(k2w, wbf, 1024L * 1024, stream);
    gemm_launch(hC, wbf, k2b, nullptr, hB, 4096, 1024, 1024, 0, stream);

    conv_launch(outw, wbf, 1024L * 1024, stream);
    gemm_launch(hB, wbf, outb, (float*)d_out, nullptr, 4096, 1024, 1024, 0, stream);
}
